// UnitAGCN_42408507081280
// MI455X (gfx1250) — matrix operand order repaired and checked by reading, not compiled, not run on hardware
//
#include <hip/hip_runtime.h>

#define NCH   64
#define TLEN  300
#define VV    25
#define TV    7500      // TLEN*VV
#define SS    3
#define INTER 16
#define KSC   4800      // INTER*TLEN
#define NB    64        // batch

typedef __attribute__((ext_vector_type(8)))  float  v8f;
typedef __attribute__((ext_vector_type(16))) __bf16 v16bf;
typedef __attribute__((ext_vector_type(8)))  __bf16 v8bf;

// k order used here for A and B (contiguous, not AMD's table order): lane<16 -> K {0..15}, lane>=16 -> K {16..31}
__device__ __forceinline__ int a_k0(int r, int hi) {
    return 2 * r + (hi ? 16 : 0);
}

__device__ __forceinline__ v8f wmma_bf16(v16bf a, v16bf b, v8f c) {
    return __builtin_amdgcn_wmma_f32_16x16x32_bf16(false, a, false, b, (short)0, c,
                                                   false, false);
}

// A-fragment: two contiguous 8xbf16 groups (K base..base+7 and base+8..base+15)
__device__ __forceinline__ v16bf make_a_frag(const __bf16* g0) {
    v8bf  lo = *(const v8bf*)g0;         // K group +0..7
    v8bf  hi = *(const v8bf*)(g0 + 8);   // K group +8..15
    v16bf f;
#pragma unroll
    for (int i = 0; i < 8; ++i) { f[i] = lo[i]; f[8 + i] = hi[i]; }
    return f;
}

__device__ __forceinline__ v16bf zero16() { v16bf z = {}; return z; }

// ------------------------------------------------ stage 0: transpose x -> xbT[T*V][C] bf16
__global__ __launch_bounds__(256) void k_transpose(const float* __restrict__ x,
                                                   __bf16* __restrict__ xbT) {
    __shared__ float tile[64][65];
    int n = blockIdx.x / 118, jt = blockIdx.x % 118;
    int j0 = jt * 64;
    const float* xn = x + (size_t)n * NCH * TV;
    __bf16*      xt = xbT + (size_t)n * TV * NCH;
    for (int i = threadIdx.x; i < 64 * 64; i += 256) {
        int c = i >> 6, jj = i & 63;
        int j       = j0 + jj;
        tile[c][jj] = (j < TV) ? xn[(size_t)c * TV + j] : 0.f;
    }
    __syncthreads();
    for (int i = threadIdx.x; i < 64 * 64; i += 256) {
        int jj = i >> 6, c = i & 63;
        int j = j0 + jj;
        if (j < TV) xt[(size_t)j * NCH + c] = (__bf16)tile[c][jj];
    }
}

// ------------------------------------------------ stage 1: fa/fb = [Wa;Wb] @ x
// both outputs stored [joint 25][k=KSC]: row-major, K-contiguous for the scores GEMM
__global__ __launch_bounds__(256) void k_fab(const __bf16* __restrict__ xbT,
                                             const float* __restrict__ Wa,
                                             const float* __restrict__ Wb,
                                             __bf16* __restrict__ fa,
                                             __bf16* __restrict__ fb) {
    __shared__ __bf16 wab[32][72];
    int n = blockIdx.x / SS, s = blockIdx.x % SS;
    int tid = threadIdx.x;
    for (int i = tid; i < 32 * 64; i += 256) {
        int row = i >> 6, c = i & 63;
        float v = (row < 16) ? Wa[(s * INTER + row) * NCH + c]
                             : Wb[(s * INTER + (row - 16)) * NCH + c];
        wab[row][c] = (__bf16)v;
    }
    __syncthreads();
    int lane = tid & 31, wid = tid >> 5;
    int hi = lane >> 4, ln = lane & 15;
    int hio8 = hi ? 8 : 0, hio16 = hi ? 16 : 0;
    const __bf16* xn    = xbT + (size_t)n * TV * NCH;
    __bf16*       fab_a = fa + (size_t)(n * SS + s) * VV * KSC;
    __bf16*       fab_b = fb + (size_t)(n * SS + s) * VV * KSC;
    for (int ct = wid; ct < (TV + 15) / 16; ct += 8) {
        int  j  = ct * 16 + ln;
        bool jv = j < TV;
        int  jc = jv ? j : (TV - 1);
        v8f acc[2];
        acc[0] = {}; acc[1] = {};
#pragma unroll
        for (int kt = 0; kt < 2; ++kt) {
            v16bf bfr = *(const v16bf*)(xn + (size_t)jc * NCH + kt * 32 + hio16);
            if (!jv) bfr = zero16();
#pragma unroll
            for (int mt = 0; mt < 2; ++mt) {
                v16bf afr = make_a_frag(&wab[16 * mt + ln][kt * 32 + hio16]);
                acc[mt]   = wmma_bf16(afr, bfr, acc[mt]);
            }
        }
        if (jv) {
            int t = j / VV, v = j % VV;
#pragma unroll
            for (int mt = 0; mt < 2; ++mt)
#pragma unroll
                for (int r = 0; r < 8; ++r) {
                    int   irow = 16 * mt + r + 8 * hi;
                    float val  = acc[mt][r];
                    if (irow < 16)
                        fab_a[(size_t)v * KSC + irow * TLEN + t] = (__bf16)val;
                    else
                        fab_b[(size_t)v * KSC + (irow - 16) * TLEN + t] = (__bf16)val;
                }
        }
    }
}

// ------------------------------------------------ stage 2: scores + softmax + adjacency
// scores[v][w] = sum_k fa[v][k]*fb[w][k]; softmax over v; Amat stored TRANSPOSED [w][v]
__global__ __launch_bounds__(128) void k_scores(const __bf16* __restrict__ fa,
                                                const __bf16* __restrict__ fb,
                                                const float* __restrict__ A,
                                                const float* __restrict__ PA,
                                                __bf16* __restrict__ amatT) {
    __shared__ float sc[32][33];
    int n = blockIdx.x / SS, s = blockIdx.x % SS;
    int tid = threadIdx.x, lane = tid & 31, wid = tid >> 5;
    int hi = lane >> 4, ln = lane & 15;
    int hio8 = hi ? 8 : 0, hio16 = hi ? 16 : 0;
    int mt = wid >> 1, nt = wid & 1;
    const __bf16* fap = fa + (size_t)(n * SS + s) * VV * KSC;
    const __bf16* fbp = fb + (size_t)(n * SS + s) * VV * KSC;
    int v = 16 * mt + ln;   // row of scores
    int w = 16 * nt + ln;   // col of scores
    int vc = (v < VV) ? v : (VV - 1);
    int wc = (w < VV) ? w : (VV - 1);
    v8f acc = {};
    for (int kt = 0; kt < KSC / 32; ++kt) {
        v16bf afr = make_a_frag(&fap[(size_t)vc * KSC + kt * 32 + hio16]);
        if (v >= VV) afr = zero16();
        v16bf bfr = *(const v16bf*)(&fbp[(size_t)wc * KSC + kt * 32 + hio16]);
        if (w >= VV) bfr = zero16();
        acc = wmma_bf16(afr, bfr, acc);
    }
#pragma unroll
    for (int r = 0; r < 8; ++r) sc[16 * mt + r + 8 * hi][16 * nt + ln] = acc[r];
    __syncthreads();
    if (tid < 32) {
        int     col = tid;   // w
        __bf16* out = amatT + (size_t)(n * SS + s) * 1024;
        if (col < VV) {
            const float scale = 1.0f / (float)KSC;
            float       mx    = -1e30f;
            for (int vv = 0; vv < VV; ++vv) mx = fmaxf(mx, sc[vv][col] * scale);
            float sum = 0.f;
            float e[VV];
            for (int vv = 0; vv < VV; ++vv) {
                e[vv] = __expf(sc[vv][col] * scale - mx);
                sum += e[vv];
            }
            float inv = 1.0f / sum;
            for (int vv = 0; vv < 32; ++vv) {
                float val = (vv < VV)
                                ? e[vv] * inv + A[(s * VV + vv) * VV + col] +
                                      PA[(s * VV + vv) * VV + col]
                                : 0.f;
                out[col * 32 + vv] = (__bf16)val;   // transposed: [w][v]
            }
        } else {
            for (int vv = 0; vv < 32; ++vv) out[col * 32 + vv] = (__bf16)0.f;
        }
    }
}

// ------------------------------------------------ stage 3: fused z = Wd@x, y += z@Amat
__global__ __launch_bounds__(256) void k_zy(const __bf16* __restrict__ xbT,
                                            const float* __restrict__ Wd,
                                            const __bf16* __restrict__ amatT,
                                            float* __restrict__ y,
                                            float* __restrict__ stats) {
    __shared__ __bf16 wd[64][72];
    __shared__ __bf16 am[32][32];    // [w][v]
    __shared__ __bf16 zl[64][408];
    int n = blockIdx.x / 19, tb = blockIdx.x % 19;
    int t0  = tb * 16;
    int tid = threadIdx.x, lane = tid & 31, wid = tid >> 5;
    int hi = lane >> 4, ln = lane & 15;
    int hio8 = hi ? 8 : 0, hio16 = hi ? 16 : 0;
    const __bf16* xn = xbT + (size_t)n * TV * NCH;
    v8f yacc[8][2];
#pragma unroll
    for (int i = 0; i < 8; ++i) { yacc[i][0] = {}; yacc[i][1] = {}; }
    for (int s = 0; s < SS; ++s) {
        for (int i = tid; i < 64 * 64; i += 256)
            wd[i >> 6][i & 63] = (__bf16)Wd[(size_t)s * 4096 + i];
        {
            const __bf16* ap = amatT + (size_t)(n * SS + s) * 1024;
            for (int i = tid; i < 1024; i += 256) am[i >> 5][i & 31] = ap[i];
        }
        __syncthreads();
        // GEMM1: z[64 x 400] = Wd[64x64] @ x_tile ; B from xbT (K-contiguous b128s)
        for (int ct = wid; ct < 25; ct += 8) {
            int  jc = ct * 16 + ln;     // 0..399 == (t-t0)*25 + v
            int  jg = t0 * VV + jc;     // global column
            bool jv = jg < TV;
            int  jgc = jv ? jg : (TV - 1);
            v8f zacc[4];
            zacc[0] = {}; zacc[1] = {}; zacc[2] = {}; zacc[3] = {};
#pragma unroll
            for (int kt = 0; kt < 2; ++kt) {
                v16bf bfr = *(const v16bf*)(xn + (size_t)jgc * NCH + kt * 32 + hio16);
                if (!jv) bfr = zero16();
#pragma unroll
                for (int mt = 0; mt < 4; ++mt) {
                    v16bf afr = make_a_frag(&wd[16 * mt + ln][kt * 32 + hio16]);
                    zacc[mt]  = wmma_bf16(afr, bfr, zacc[mt]);
                }
            }
#pragma unroll
            for (int mt = 0; mt < 4; ++mt)
#pragma unroll
                for (int r = 0; r < 8; ++r)
                    zl[16 * mt + r + 8 * hi][jc] = (__bf16)zacc[mt][r];
        }
        __syncthreads();
        // GEMM2: y[(o,t), w] += z[(o,t), v] @ Amat[v][w]  (K=V padded to 32)
        v16bf b2[2];
#pragma unroll
        for (int nt = 0; nt < 2; ++nt)
            b2[nt] = *(const v16bf*)(&am[16 * nt + ln][hio16]);
#pragma unroll
        for (int mi = 0; mi < 8; ++mi) {
            int   o = wid * 8 + mi;
            v16bf a2;
#pragma unroll
            for (int r = 0; r < 8; ++r) {
                int v0        = a_k0(r, hi);
                int idx       = ln * VV + v0;   // <= 406 < 408, always in-bounds
                __bf16 e0     = zl[o][idx];
                __bf16 e1     = zl[o][idx + 1];
                a2[2 * r]     = (v0 < VV) ? e0 : (__bf16)0.f;
                a2[2 * r + 1] = (v0 + 1 < VV) ? e1 : (__bf16)0.f;
            }
            yacc[mi][0] = wmma_bf16(a2, b2[0], yacc[mi][0]);
            yacc[mi][1] = wmma_bf16(a2, b2[1], yacc[mi][1]);
        }
        __syncthreads();
    }
    // write y + BatchNorm partial stats
#pragma unroll
    for (int mi = 0; mi < 8; ++mi) {
        int   o    = wid * 8 + mi;
        float lsum = 0.f, lsq = 0.f;
#pragma unroll
        for (int nt = 0; nt < 2; ++nt) {
#pragma unroll
            for (int r = 0; r < 8; ++r) {
                int   tl  = r + 8 * hi;
                int   t   = t0 + tl;
                int   w   = 16 * nt + ln;
                float val = yacc[mi][nt][r];
                if (t < TLEN && w < VV) {
                    y[(((size_t)n * NCH + o) * TLEN + t) * VV + w] = val;
                    lsum += val;
                    lsq += val * val;
                }
            }
        }
#pragma unroll
        for (int off = 16; off > 0; off >>= 1) {
            lsum += __shfl_down(lsum, off);
            lsq += __shfl_down(lsq, off);
        }
        if (lane == 0) {
            atomicAdd(&stats[o], lsum);
            atomicAdd(&stats[64 + o], lsq);
        }
    }
}

// ------------------------------------------------ BN finalize + epilogue
__global__ void k_init_stats(float* stats) { stats[threadIdx.x] = 0.f; }

__global__ void k_bn_finalize(float* stats) {
    int o = threadIdx.x;
    if (o < 64) {
        const float cnt  = (float)NB * (float)TLEN * (float)VV;
        float       mean = stats[o] / cnt;
        float       var  = stats[64 + o] / cnt - mean * mean;
        stats[128 + o]   = mean;
        stats[192 + o]   = rsqrtf(var + 1e-5f);
    }
}

__global__ __launch_bounds__(256) void k_out(const float* __restrict__ y,
                                             const float* __restrict__ x,
                                             const float* __restrict__ gamma,
                                             const float* __restrict__ beta,
                                             const float* __restrict__ stats,
                                             float* __restrict__ out, size_t nelem) {
    size_t i = (size_t)blockIdx.x * 256 + threadIdx.x;
    if (i >= nelem) return;
    int   o   = (int)((i / (size_t)TV) % NCH);
    float val = y[i];
    float r   = gamma[o] * (val - stats[128 + o]) * stats[192 + o] + beta[o] + x[i];
    out[i]    = fmaxf(r, 0.f);
}

// ------------------------------------------------ launcher
extern "C" void kernel_launch(void* const* d_in, const int* in_sizes, int n_in,
                              void* d_out, int out_size, void* d_ws, size_t ws_size,
                              hipStream_t stream) {
    const float* x     = (const float*)d_in[0];
    const float* A     = (const float*)d_in[1];
    const float* PA    = (const float*)d_in[2];
    const float* Wa    = (const float*)d_in[3];
    const float* Wb    = (const float*)d_in[4];
    const float* Wd    = (const float*)d_in[5];
    const float* gamma = (const float*)d_in[6];
    const float* beta  = (const float*)d_in[7];
    float*       out   = (float*)d_out;

    const size_t NX = (size_t)NB * NCH * TV;   // 30,720,000

    char* p     = (char*)d_ws;
    auto  alloc = [&](size_t bytes) -> void* {
        void* r = (void*)p;
        p += (bytes + 255) & ~(size_t)255;
        return r;
    };
    __bf16* xbT   = (__bf16*)alloc(NX * 2);
    __bf16* fa    = (__bf16*)alloc((size_t)NB * SS * VV * KSC * 2);
    __bf16* fb    = (__bf16*)alloc((size_t)NB * SS * VV * KSC * 2);
    __bf16* amatT = (__bf16*)alloc((size_t)NB * SS * 1024 * 2);
    float*  y     = (float*)alloc(NX * 4);
    float*  stats = (float*)alloc(256 * 4);

    k_init_stats<<<1, 256, 0, stream>>>(stats);
    k_transpose<<<NB * 118, 256, 0, stream>>>(x, xbT);
    k_fab<<<NB * SS, 256, 0, stream>>>(xbT, Wa, Wb, fa, fb);
    k_scores<<<NB * SS, 128, 0, stream>>>(fa, fb, A, PA, amatT);
    k_zy<<<NB * 19, 256, 0, stream>>>(xbT, Wd, amatT, y, stats);
    k_bn_finalize<<<1, 64, 0, stream>>>(stats);
    k_out<<<(int)((NX + 255) / 256), 256, 0, stream>>>(y, x, gamma, beta, stats, out,
                                                       NX);
}
